// TabNetEncoder_13563506720926
// MI455X (gfx1250) — matrix operand order repaired and checked by reading, not compiled, not run on hardware
//
#include <hip/hip_runtime.h>

#define NTHREADS 256
#define TM 32
#define NSTEPS_ 3
#define RELAXC 1.3f
#define BNEPS 0.001f
#define SQH 0.70710678118654752440f

typedef __attribute__((ext_vector_type(16))) __bf16 v16bf;
typedef __attribute__((ext_vector_type(8)))  float  v8f;

#if defined(__HIP_DEVICE_COMPILE__) && \
    __has_builtin(__builtin_amdgcn_global_load_async_to_lds_b128) && \
    __has_builtin(__builtin_amdgcn_s_wait_asynccnt)
#define USE_ASYNC_LDS 1
typedef int v4i_async __attribute__((vector_size(16)));
typedef __attribute__((address_space(1))) v4i_async* g_v4i_ptr;
typedef __attribute__((address_space(3))) v4i_async* l_v4i_ptr;
#else
#define USE_ASYNC_LDS 0
#endif

union ABfrag { v16bf v; uint4 u[2]; };

// A-matrix 16x32 bf16 fragment (per ISA: lanes0-15 K=0..7,16..23 ; lanes16-31 K=8..15,24..31)
__device__ inline v16bf load_a_frag(const __bf16* Abuf, int K, int rowbase, int kb, int lane) {
    int mrow = lane & 15;
    int off  = (lane < 16) ? 0 : 8;
    const __bf16* p = Abuf + (rowbase + mrow) * K + kb + off;
    ABfrag f;
    f.u[0] = *reinterpret_cast<const uint4*>(p);
    f.u[1] = *reinterpret_cast<const uint4*>(p + 16);
    return f.v;
}

// B-matrix 32x16 bf16 fragment (lanes0-15: K=0..7,16..23 ; lanes16-31: K=8..15,24..31, col = lane%16)
// wb holds W^T rows: wb[n*K + k]
__device__ inline v16bf load_b_frag(const __bf16* wb, int K, int nlocal, int kb, int lane) {
    int off = (lane < 16) ? 0 : 8;
    const __bf16* p = wb + nlocal * K + kb + off;
    ABfrag f;
    f.u[0] = *reinterpret_cast<const uint4*>(p);
    f.u[1] = *reinterpret_cast<const uint4*>(p + 16);
    return f.v;
}

// Z[0:32, n0:n0+NR] = A[0:32, 0:K] * W^T(rows n0..)^T + bias ; Z stride 256 f32 in LDS
template<int TPW>
__device__ void gemm_tile(const __bf16* Abuf, int K, const __bf16* wb, int n0,
                          const float* biasG, float* Zout) {
    const int wave = threadIdx.x >> 5;
    const int lane = threadIdx.x & 31;
    v8f zero = {};
    v8f acc[TPW];
#pragma unroll
    for (int t = 0; t < TPW; ++t) acc[t] = zero;

    for (int kb = 0; kb < K; kb += 32) {
        if constexpr (TPW == 1) {
            int mt = wave & 1, nt = wave >> 1;
            v16bf a = load_a_frag(Abuf, K, mt * 16, kb, lane);
            v16bf b = load_b_frag(wb, K, nt * 16 + (lane & 15), kb, lane);
            acc[0] = __builtin_amdgcn_wmma_f32_16x16x32_bf16(false, a, false, b,
                                                             (short)0, acc[0], false, false);
        } else {
            v16bf a0 = load_a_frag(Abuf, K, 0, kb, lane);
            v16bf a1 = load_a_frag(Abuf, K, 16, kb, lane);
#pragma unroll
            for (int t = 0; t < TPW; t += 2) {
                int tid2 = wave * TPW + t;
                int nt = tid2 >> 1;
                v16bf b = load_b_frag(wb, K, nt * 16 + (lane & 15), kb, lane);
                acc[t]     = __builtin_amdgcn_wmma_f32_16x16x32_bf16(false, a0, false, b,
                                                                     (short)0, acc[t], false, false);
                acc[t + 1] = __builtin_amdgcn_wmma_f32_16x16x32_bf16(false, a1, false, b,
                                                                     (short)0, acc[t + 1], false, false);
            }
        }
    }
#pragma unroll
    for (int t = 0; t < TPW; ++t) {
        int tid2 = wave * TPW + t;
        int mt = (TPW == 1) ? (wave & 1)  : (t & 1);
        int nt = (TPW == 1) ? (wave >> 1) : (tid2 >> 1);
        int n = n0 + nt * 16 + (lane & 15);
        float bz = biasG[n];
        int mbase = mt * 16 + ((lane < 16) ? 0 : 8);
#pragma unroll
        for (int r = 0; r < 8; ++r)
            Zout[(mbase + r) * 256 + n] = acc[t][r] + bz;
    }
}

// One logical matmul stage: stream W^T slices through a 32KB LDS buffer.
__device__ void matmul_stage(const __bf16* __restrict__ WTg, const float* __restrict__ biasG,
                             const __bf16* Abuf, int K, int N, float* Zout, __bf16* wbuf) {
    int NR = 16384 / K; if (NR > N) NR = N;
    int tpw = (2 * (NR >> 4)) >> 3;   // 1, 2 or 4
    for (int n0 = 0; n0 < N; n0 += NR) {
        const uint4* src = reinterpret_cast<const uint4*>(WTg + (size_t)n0 * K);
        uint4* dst = reinterpret_cast<uint4*>(wbuf);
        int chunks = (NR * K) >> 3;
#if USE_ASYNC_LDS
        for (int i = threadIdx.x; i < chunks; i += NTHREADS) {
            __builtin_amdgcn_global_load_async_to_lds_b128(
                (g_v4i_ptr)(unsigned long long)(src + i),
                (l_v4i_ptr)(unsigned)(unsigned long long)(dst + i),
                0, 0);
        }
        __builtin_amdgcn_s_wait_asynccnt(0);
#else
        for (int i = threadIdx.x; i < chunks; i += NTHREADS) dst[i] = src[i];
#endif
        __syncthreads();
        if (tpw == 1)      gemm_tile<1>(Abuf, K, wbuf, n0, biasG, Zout);
        else if (tpw == 2) gemm_tile<2>(Abuf, K, wbuf, n0, biasG, Zout);
        else               gemm_tile<4>(Abuf, K, wbuf, n0, biasG, Zout);
        __syncthreads();
    }
}

__global__ __launch_bounds__(NTHREADS)
void tabnet_main(const float* __restrict__ inputs,
                 const float* __restrict__ in_g, const float* __restrict__ in_b,
                 const float* __restrict__ in_m, const float* __restrict__ in_v,
                 const __bf16* __restrict__ shW0T, const __bf16* __restrict__ shW1T,
                 const __bf16* __restrict__ depT,  const __bf16* __restrict__ atT,
                 const __bf16* __restrict__ WoutT,
                 const float* __restrict__ b_sh0, const float* __restrict__ b_sh1,
                 const float* __restrict__ b_dep, const float* __restrict__ b_at,
                 const float* __restrict__ b_zero,
                 float* __restrict__ out) {
    __shared__ __align__(16) __bf16 wbuf[16384];       // 32 KB rotating weight slice (W^T)
    __shared__ __align__(16) __bf16 featb[TM * 256];   // 16 KB bf16 features
    __shared__ __align__(16) float  priorb[TM * 256];  // 32 KB
    __shared__ __align__(16) float  zbuf[TM * 256];    // 32 KB (logits / GEMM output)
    __shared__ __align__(16) __bf16 xin[TM * 256];     // 16 KB GEMM A-operand (bf16, K-packed)
    __shared__ __align__(16) float  xbuf[TM * 128];    // 16 KB running activation x
    __shared__ __align__(16) float  oagg[TM * 64];     //  8 KB output aggregate

    const int tid = threadIdx.x;
    const long row0 = (long)blockIdx.x * TM;

    // ---- features = BN(inputs); prior = 1; oagg = 0 ----
    for (int i = tid; i < TM * 256; i += NTHREADS) {
        int r = i >> 8, c = i & 255;
        float x = inputs[(row0 + r) * 256 + c];
        float s = in_g[c] * rsqrtf(in_v[c] + BNEPS);
        featb[i]  = (__bf16)((x - in_m[c]) * s + in_b[c]);
        priorb[i] = 1.0f;
    }
    for (int i = tid; i < TM * 64; i += NTHREADS) oagg[i] = 0.0f;
    __syncthreads();

    auto glu = [&](bool residual) {
        for (int i = tid; i < TM * 128; i += NTHREADS) {
            int r = i >> 7, c = i & 127;
            float z1 = zbuf[r * 256 + c];
            float z2 = zbuf[r * 256 + 128 + c];
            float h  = z1 / (1.0f + __expf(-z2));
            float x  = residual ? SQH * (xbuf[i] + h) : h;
            xbuf[i] = x;
            xin[i]  = (__bf16)x;   // K=128 packed
        }
        __syncthreads();
    };
    auto shared_block = [&](const __bf16* A0) {
        matmul_stage(shW0T, b_sh0, A0, 256, 256, zbuf, wbuf);
        glu(false);
        matmul_stage(shW1T, b_sh1, xin, 128, 256, zbuf, wbuf);
        glu(true);
    };
    auto dep_block = [&](int t) {
        for (int i = 0; i < 2; ++i) {
            int idx = t * 2 + i;
            matmul_stage(depT + (size_t)idx * (256 * 128), b_dep + idx * 256,
                         xin, 128, 256, zbuf, wbuf);
            glu(true);
        }
    };

    // ---- initial shared + dep block ----
    shared_block(featb);
    dep_block(0);
    for (int i = tid; i < TM * 64; i += NTHREADS) {       // a = x[:,64:] -> xin (K=64)
        int r = i >> 6, c = i & 63;
        xin[i] = (__bf16)xbuf[r * 128 + 64 + c];
    }
    __syncthreads();

    // ---- decision steps ----
    for (int s = 0; s < NSTEPS_; ++s) {
        matmul_stage(atT + (size_t)s * (256 * 64), b_at + s * 256, xin, 64, 256, zbuf, wbuf);

        // sparsemax(prior * logits) via Michelot fixed-point; 8 lanes per row
        {
            int r = tid >> 3, j = tid & 7;
            int base = r * 256 + j * 32;
            float zv[32];
            float S = 0.0f;
#pragma unroll
            for (int q = 0; q < 32; ++q) {
                float z = priorb[base + q] * zbuf[base + q];
                zv[q] = z; S += z;
            }
#pragma unroll
            for (int d = 1; d < 8; d <<= 1) S += __shfl_xor(S, d, 8);
            float tau = (S - 1.0f) * (1.0f / 256.0f);
            for (int it = 0; it < 16; ++it) {
                float ss = 0.0f, cnt = 0.0f;
#pragma unroll
                for (int q = 0; q < 32; ++q)
                    if (zv[q] > tau) { ss += zv[q]; cnt += 1.0f; }
#pragma unroll
                for (int d = 1; d < 8; d <<= 1) {
                    ss  += __shfl_xor(ss,  d, 8);
                    cnt += __shfl_xor(cnt, d, 8);
                }
                tau = (ss - 1.0f) / cnt;
            }
#pragma unroll
            for (int q = 0; q < 32; ++q) {
                float m = fmaxf(zv[q] - tau, 0.0f);
                priorb[base + q] *= (RELAXC - m);
                xin[base + q] = (__bf16)(m * (float)featb[base + q]);  // K=256 packed
            }
        }
        __syncthreads();

        shared_block(xin);
        dep_block(s + 1);

        for (int i = tid; i < TM * 64; i += NTHREADS) {    // agg relu + a -> xin
            int r = i >> 6, c = i & 63;
            oagg[i] += fmaxf(xbuf[r * 128 + c], 0.0f);
            xin[i] = (__bf16)xbuf[r * 128 + 64 + c];
        }
        __syncthreads();
    }

    // ---- out = oagg @ Wout ----
    for (int i = tid; i < TM * 64; i += NTHREADS) xin[i] = (__bf16)oagg[i];
    __syncthreads();
    matmul_stage(WoutT, b_zero, xin, 64, 64, zbuf, wbuf);
    for (int i = tid; i < TM * 64; i += NTHREADS) {
        int r = i >> 6, c = i & 63;
        out[(row0 + r) * 64 + c] = zbuf[r * 256 + c];
    }
}

// Fold BN into weights: WT[n][k] = bf16(W[k][n] * g[n]*rsqrt(v[n]+eps)); bias[n] = b[n]-m[n]*scale
__global__ void prep_wt(const float* __restrict__ W, const float* __restrict__ g,
                        const float* __restrict__ b, const float* __restrict__ m,
                        const float* __restrict__ v,
                        __bf16* __restrict__ WT, float* __restrict__ bias, int K, int N) {
    int i = blockIdx.x * blockDim.x + threadIdx.x;
    if (i >= N * K) return;
    int n = i / K, k = i % K;
    float s = 1.0f, bb = 0.0f;
    if (g) { float sc = g[n] * rsqrtf(v[n] + BNEPS); s = sc; bb = b[n] - m[n] * sc; }
    WT[i] = (__bf16)(W[(size_t)k * N + n] * s);
    if (bias && k == 0) bias[n] = bb;
}

extern "C" void kernel_launch(void* const* d_in, const int* in_sizes, int n_in,
                              void* d_out, int out_size, void* d_ws, size_t ws_size,
                              hipStream_t stream) {
    (void)in_sizes; (void)n_in; (void)out_size; (void)ws_size;
    const float* inputs = (const float*)d_in[0];
    const float* in_g  = (const float*)d_in[1];
    const float* in_b  = (const float*)d_in[2];
    const float* in_m  = (const float*)d_in[3];
    const float* in_v  = (const float*)d_in[4];
    const float* sh_W0 = (const float*)d_in[5];
    const float* sh_W1 = (const float*)d_in[6];
    const float* sh_g  = (const float*)d_in[7];
    const float* sh_b  = (const float*)d_in[8];
    const float* sh_m  = (const float*)d_in[9];
    const float* sh_v  = (const float*)d_in[10];
    const float* dep_W = (const float*)d_in[11];
    const float* dep_g = (const float*)d_in[12];
    const float* dep_b = (const float*)d_in[13];
    const float* dep_m = (const float*)d_in[14];
    const float* dep_v = (const float*)d_in[15];
    const float* at_W  = (const float*)d_in[16];
    const float* at_g  = (const float*)d_in[17];
    const float* at_b  = (const float*)d_in[18];
    const float* at_m  = (const float*)d_in[19];
    const float* at_v  = (const float*)d_in[20];
    const float* Wout  = (const float*)d_in[21];

    char* ws = (char*)d_ws;
    size_t off = 0;
    __bf16* shW0T = (__bf16*)(ws + off); off += (size_t)256 * 256 * 2;
    __bf16* shW1T = (__bf16*)(ws + off); off += (size_t)256 * 128 * 2;
    __bf16* depT  = (__bf16*)(ws + off); off += (size_t)8 * 256 * 128 * 2;
    __bf16* atT   = (__bf16*)(ws + off); off += (size_t)3 * 256 * 64 * 2;
    __bf16* WoutT = (__bf16*)(ws + off); off += (size_t)64 * 64 * 2;
    float* b_sh0  = (float*)(ws + off); off += 256 * 4;
    float* b_sh1  = (float*)(ws + off); off += 256 * 4;
    float* b_dep  = (float*)(ws + off); off += 8 * 256 * 4;
    float* b_at   = (float*)(ws + off); off += 3 * 256 * 4;
    float* b_zero = (float*)(ws + off); off += 256 * 4;

    auto prep = [&](const float* W, const float* g, const float* b2, const float* m,
                    const float* v, __bf16* WT, float* bias, int K, int N) {
        int total = N * K;
        prep_wt<<<(total + 255) / 256, 256, 0, stream>>>(W, g, b2, m, v, WT, bias, K, N);
    };
    prep(sh_W0, sh_g,       sh_b,       sh_m,       sh_v,       shW0T, b_sh0, 256, 256);
    prep(sh_W1, sh_g + 256, sh_b + 256, sh_m + 256, sh_v + 256, shW1T, b_sh1, 128, 256);
    for (int idx = 0; idx < 8; ++idx)
        prep(dep_W + (size_t)idx * 128 * 256, dep_g + idx * 256, dep_b + idx * 256,
             dep_m + idx * 256, dep_v + idx * 256,
             depT + (size_t)idx * 256 * 128, b_dep + idx * 256, 128, 256);
    for (int s = 0; s < 3; ++s)
        prep(at_W + (size_t)s * 64 * 256, at_g + s * 256, at_b + s * 256,
             at_m + s * 256, at_v + s * 256,
             atT + (size_t)s * 256 * 64, b_at + s * 256, 64, 256);
    prep(Wout, nullptr, nullptr, nullptr, nullptr, WoutT, b_zero, 64, 64);  // zero bias

    tabnet_main<<<131072 / TM, NTHREADS, 0, stream>>>(
        inputs, in_g, in_b, in_m, in_v,
        shW0T, shW1T, depT, atT, WoutT,
        b_sh0, b_sh1, b_dep, b_at, b_zero, (float*)d_out);
}
